// deform_dconv_48996986913001
// MI455X (gfx1250) — matrix operand order repaired and checked by reading, not compiled, not run on hardware
//
#include <hip/hip_runtime.h>

// Deformable conv2d (B=8, CIN=64, COUT=64, K=3, S=4, H=W=64, stride=1, pad=1, dil=1)
// fused bilinear-gather + WMMA GEMM for gfx1250 (wave32, 16x16x32 f16 WMMA, f32 accum).
// Round 1: merged the two cin-halves of each 3x3 tap into one loop iteration so the
// offset loads + bilinear coefficient math are computed once per tap (was twice),
// and each tap iteration now issues 8 v_wmma_f32_16x16x32_f16.

typedef __attribute__((ext_vector_type(16))) _Float16 v16h;
typedef __attribute__((ext_vector_type(8)))  float    v8f;
typedef __attribute__((ext_vector_type(4)))  float    v4f;

namespace {
constexpr int B_    = 8;
constexpr int CIN_  = 64;
constexpr int COUT_ = 64;
constexpr int KK_   = 9;            // 3x3 taps
constexpr int S_    = 4;
constexpr int H_    = 64;
constexpr int W_    = 64;
constexpr int KTOT_ = KK_ * CIN_;   // 576 = GEMM K
constexpr int CSTR_ = S_ * H_ * W_; // 16384 floats: channel stride in input/offset/output
constexpr int THREADS_ = 128;       // 4 waves per block
constexpr int PIXB_    = 64;        // pixels (M rows) per block = one y-row
constexpr int NPIX_ = B_ * S_ * H_ * W_; // 131072 = GEMM M
}

__global__ __launch_bounds__(THREADS_)
void deform_conv_wmma(const float* __restrict__ inp,
                      const float* __restrict__ off,
                      const float* __restrict__ wgt,
                      const float* __restrict__ bias,
                      float* __restrict__ out)
{
    // Weight staged as W[cout][kidx*CIN + cin] in f16: 64*576*2 = 72 KB LDS
    // (fits 320 KB/WGP at 4 blocks per WGP).
    __shared__ _Float16 wlds[COUT_ * KTOT_];

    const int tid  = threadIdx.x;
    const int lane = tid & 31;
    const int wave = tid >> 5;

    // ---- cooperative weight stage (global f32 -> LDS f16, transposed cin<->kk) ----
    for (int e = tid; e < COUT_ * KTOT_; e += THREADS_) {
        const int kk   = e % KK_;
        const int cin  = (e / KK_) % CIN_;
        const int cout = e / (KK_ * CIN_);
        wlds[cout * KTOT_ + kk * CIN_ + cin] = (_Float16)wgt[e];
    }
    __syncthreads();

    // ---- this lane's identity in the WMMA fragments ----
    const int m  = lane & 15;   // A-matrix row (pixel within wave tile)
    const int hb = lane >> 4;   // half-select per ISA 16-bit fragment layout
    const int n  = lane & 15;   // B/C/D column (cout within N-tile)

    const int p = blockIdx.x * PIXB_ + wave * 16 + m;   // global pixel (b,s,y,x)
    const int x = p & 63;
    const int y = (p >> 6) & 63;
    const int s = (p >> 12) & 3;
    const int b = p >> 14;

    const int inpB = (b * CIN_ * S_ + s) * (H_ * W_);                    // + c*CSTR_
    const int offB = (b * (2 * KK_) * S_ + s) * (H_ * W_) + y * W_ + x;  // + ch*CSTR_

    v8f acc[4] = {};   // 4 N-tiles of 16 couts, f32 accumulators

    // Tap loop: 9 taps; each tap contributes K=64 (two 16x16x32 WMMA chunks).
    for (int ky = 0; ky < 3; ++ky) {
        for (int kx = 0; kx < 3; ++kx) {
            const int kidx = ky * 3 + kx;

            // Bilinear sample position for (pixel m, tap kidx);
            // shared by all 32 channels this lane contributes.
            const float dy = off[offB + (2 * kidx    ) * CSTR_];
            const float dx = off[offB + (2 * kidx + 1) * CSTR_];
            const float py = (float)(y - 1 + ky) + dy;
            const float px = (float)(x - 1 + kx) + dx;
            const float y0f = floorf(py);
            const float x0f = floorf(px);
            const float wy  = py - y0f;
            const float wx  = px - x0f;
            const int   y0  = (int)y0f, x0 = (int)x0f;
            const int   y1  = y0 + 1,   x1 = x0 + 1;
            const float vy0 = (y0 >= 0 && y0 < H_) ? 1.0f : 0.0f;
            const float vy1 = (y1 >= 0 && y1 < H_) ? 1.0f : 0.0f;
            const float vx0 = (x0 >= 0 && x0 < W_) ? 1.0f : 0.0f;
            const float vx1 = (x1 >= 0 && x1 < W_) ? 1.0f : 0.0f;
            const float c00 = (1.0f - wy) * (1.0f - wx) * vy0 * vx0;
            const float c01 = (1.0f - wy) * wx          * vy0 * vx1;
            const float c10 = wy          * (1.0f - wx) * vy1 * vx0;
            const float c11 = wy          * wx          * vy1 * vx1;
            const int iy0 = min(max(y0, 0), H_ - 1);
            const int iy1 = min(max(y1, 0), H_ - 1);
            const int ix0 = min(max(x0, 0), W_ - 1);
            const int ix1 = min(max(x1, 0), W_ - 1);
            const int o00 = iy0 * W_ + ix0, o01 = iy0 * W_ + ix1;
            const int o10 = iy1 * W_ + ix0, o11 = iy1 * W_ + ix1;

            // Build two A fragments (16x32 f16): cin 0..31 and 32..63.
            // Element i of v16h = VGPR i/2 half i%2:
            //   klocal = hb*16 + i (contiguous, matches B)
            v16h a0, a1;
            #pragma unroll
            for (int i = 0; i < 16; ++i) {
                const int cl = hb * 16 + i;
                const float* b0p = inp + inpB + cl * CSTR_;
                const float* b1p = b0p + 32 * CSTR_;
                a0[i] = (_Float16)(c00 * b0p[o00] + c01 * b0p[o01]
                                 + c10 * b0p[o10] + c11 * b0p[o11]);
                a1[i] = (_Float16)(c00 * b1p[o00] + c01 * b1p[o01]
                                 + c10 * b1p[o10] + c11 * b1p[o11]);
            }

            // B fragments (32x16 f16): lane holds column n, K = hb*16 + 0..15 contiguous.
            const int kb0 = (kidx * 2) * 32 + hb * 16;  // cin half 0
            const int kb1 = kb0 + 32;                   // cin half 1
            #pragma unroll
            for (int nt = 0; nt < 4; ++nt) {
                const v16h bf0 = *(const v16h*)&wlds[(nt * 16 + n) * KTOT_ + kb0];
                acc[nt] = __builtin_amdgcn_wmma_f32_16x16x32_f16(
                    false, a0, false, bf0, (short)0, acc[nt], false, false);
            }
            #pragma unroll
            for (int nt = 0; nt < 4; ++nt) {
                const v16h bf1 = *(const v16h*)&wlds[(nt * 16 + n) * KTOT_ + kb1];
                acc[nt] = __builtin_amdgcn_wmma_f32_16x16x32_f16(
                    false, a1, false, bf1, (short)0, acc[nt], false, false);
            }
        }
    }

    // ---- epilogue: D fragment (VGPR r, lane) -> pixel x = wave*16 + hb*8 + r, cout = nt*16+n ----
    const int outRow = (b * COUT_ * S_ + s) * (H_ * W_) + y * W_;  // + cout*CSTR_ + x
    const int xb = wave * 16 + hb * 8;
    #pragma unroll
    for (int nt = 0; nt < 4; ++nt) {
        const int cout = nt * 16 + n;
        const float bv = bias[cout];
        float* o = out + outRow + cout * CSTR_ + xb;
        v4f lo = { acc[nt][0] + bv, acc[nt][1] + bv, acc[nt][2] + bv, acc[nt][3] + bv };
        v4f hi = { acc[nt][4] + bv, acc[nt][5] + bv, acc[nt][6] + bv, acc[nt][7] + bv };
        *(v4f*)(o)     = lo;
        *(v4f*)(o + 4) = hi;
    }
}

extern "C" void kernel_launch(void* const* d_in, const int* in_sizes, int n_in,
                              void* d_out, int out_size, void* d_ws, size_t ws_size,
                              hipStream_t stream)
{
    const float* inp  = (const float*)d_in[0];   // (8, 64, 4, 64, 64) f32
    const float* off  = (const float*)d_in[1];   // (8, 18, 4, 64, 64) f32
    const float* wgt  = (const float*)d_in[2];   // (64, 64, 3, 3)     f32
    const float* bias = (const float*)d_in[3];   // (64,)              f32
    float* out = (float*)d_out;                  // (8, 64, 4, 64, 64) f32

    const int nblocks = NPIX_ / PIXB_;           // 2048
    deform_conv_wmma<<<dim3(nblocks), dim3(THREADS_), 0, stream>>>(inp, off, wgt, bias, out);

    (void)in_sizes; (void)n_in; (void)out_size; (void)d_ws; (void)ws_size;
}
